// DeepseekV3MLPMoEModel_54090818126359
// MI455X (gfx1250) — matrix operand order repaired and checked by reading, not compiled, not run on hardware
//
// MI455X (gfx1250) fused transformer forward: bf16 WMMA GEMMs + flash attention.
// fp32 in memory, bf16 v_wmma_f32_16x16x32_bf16 compute, f32 accumulate.

#include <hip/hip_runtime.h>

#define DEV_INLINE __device__ __forceinline__

typedef __bf16 bf16;
typedef __attribute__((ext_vector_type(8)))  __bf16 v8bf;
typedef __attribute__((ext_vector_type(16))) __bf16 v16bf;
typedef __attribute__((ext_vector_type(8)))  float  v8f;

// Model constants (from reference)
constexpr int CB = 2, CS = 2048, CD = 1024, CH = 16, CDH = 64, CF = 2048, CE = 8;
constexpr int MTOK = CB * CS;   // 4096 tokens
constexpr int CV = 32000;

// ---------------------------------------------------------------------------
// WMMA fragment helpers (ISA 7.12.2 layouts, wave32)
// A 16x32 bf16: lane m = lane&15; k-chunks {ko..ko+7, ko+16..ko+23}, ko = lane<16?0:8
// B 32x16 bf16: lane n = lane&15; k-chunks {ko..ko+7, ko+16..ko+23}, ko = lane<16?0:8
// C/D 16x16 f32: n = lane&15, m = (lane>>4)*8 + vgpr
// ---------------------------------------------------------------------------
DEV_INLINE v16bf frag_a(const bf16* rowp, int lane) {
  const bf16* p = rowp + ((lane & 16) ? 8 : 0);
  v8bf lo = *(const v8bf*)(p);
  v8bf hi = *(const v8bf*)(p + 16);
  return __builtin_shufflevector(lo, hi, 0,1,2,3,4,5,6,7,8,9,10,11,12,13,14,15);
}
DEV_INLINE v16bf frag_b(const bf16* rowp, int lane) {
  const bf16* p = rowp + ((lane & 16) ? 8 : 0);
  v8bf lo = *(const v8bf*)(p);
  v8bf hi = *(const v8bf*)(p + 16);
  return __builtin_shufflevector(lo, hi, 0,1,2,3,4,5,6,7,8,9,10,11,12,13,14,15);
}
DEV_INLINE v8f wmma_bf16(v16bf a, v16bf b, v8f c) {
  return __builtin_amdgcn_wmma_f32_16x16x32_bf16(false, a, false, b, (short)0, c,
                                                 false, false);
}

enum { ACT_NONE = 0, ACT_SILU = 1, ACT_RELU = 2, ACT_MOE = 3 };

// ---------------------------------------------------------------------------
// Generic GEMM: C[M,N] = act(A[M,K] * W[N,K]^T + bias), fp32 <-> bf16 WMMA.
// Block tile 128x128, BK=64, 256 threads = 8 waves, each wave 32x64 (2x4 tiles).
// Requires M%128==0, N%128==0, K%64==0 (true for all call sites).
// ---------------------------------------------------------------------------
template <int ACT>
__global__ __launch_bounds__(256) void gemm_k(
    const float* __restrict__ A, const float* __restrict__ W,
    const float* __restrict__ bias, float* __restrict__ C,
    int M, int N, int K,
    const float* __restrict__ gatew, int gstride) {
  (void)M;
  constexpr int LT = 72;  // LDS row stride in bf16: 144B, bank-conflict-free, 16B aligned
  __shared__ __align__(16) bf16 As[128 * LT];
  __shared__ __align__(16) bf16 Ws[128 * LT];
  const int tid = threadIdx.x, lane = tid & 31, wid = tid >> 5;
  const int wm = (wid & 3) * 32, wn = (wid >> 2) * 64;
  const int bm = blockIdx.y * 128, bn = blockIdx.x * 128;

  v8f acc[2][4];
#pragma unroll
  for (int i = 0; i < 2; ++i)
#pragma unroll
    for (int j = 0; j < 4; ++j)
#pragma unroll
      for (int r = 0; r < 8; ++r) acc[i][j][r] = 0.f;

  for (int k0 = 0; k0 < K; k0 += 64) {
    // Stage 128x64 fp32 tiles -> bf16 LDS (8 float4 per thread per matrix)
#pragma unroll
    for (int i = 0; i < 8; ++i) {
      int idx = tid + i * 256;
      int r = idx >> 4, c = (idx & 15) << 2;
      float4 av = *(const float4*)(A + (size_t)(bm + r) * K + k0 + c);
      bf16* da = &As[r * LT + c];
      da[0] = (bf16)av.x; da[1] = (bf16)av.y; da[2] = (bf16)av.z; da[3] = (bf16)av.w;
      float4 wv = *(const float4*)(W + (size_t)(bn + r) * K + k0 + c);
      bf16* dw = &Ws[r * LT + c];
      dw[0] = (bf16)wv.x; dw[1] = (bf16)wv.y; dw[2] = (bf16)wv.z; dw[3] = (bf16)wv.w;
    }
    if (k0 + 64 < K) {  // global_prefetch_b8 next K tile
      __builtin_prefetch(A + (size_t)(bm + (tid >> 1)) * K + k0 + 64, 0, 1);
      __builtin_prefetch(W + (size_t)(bn + (tid >> 1)) * K + k0 + 64, 0, 1);
    }
    __syncthreads();
#pragma unroll
    for (int kk = 0; kk < 64; kk += 32) {
      v16bf af[2], wf[4];
#pragma unroll
      for (int mt = 0; mt < 2; ++mt)
        af[mt] = frag_a(&As[(wm + mt * 16 + (lane & 15)) * LT + kk], lane);
#pragma unroll
      for (int nt = 0; nt < 4; ++nt)
        wf[nt] = frag_b(&Ws[(wn + nt * 16 + (lane & 15)) * LT + kk], lane);
#pragma unroll
      for (int mt = 0; mt < 2; ++mt)
#pragma unroll
        for (int nt = 0; nt < 4; ++nt)
          acc[mt][nt] = wmma_bf16(af[mt], wf[nt], acc[mt][nt]);
    }
    __syncthreads();
  }

  const int nl = lane & 15, mr = (lane >> 4) * 8;
#pragma unroll
  for (int mt = 0; mt < 2; ++mt)
#pragma unroll
    for (int nt = 0; nt < 4; ++nt) {
      int gn = bn + wn + nt * 16 + nl;
      float bb = bias ? bias[gn] : 0.f;
#pragma unroll
      for (int r = 0; r < 8; ++r) {
        int gm = bm + wm + mt * 16 + mr + r;
        float v = acc[mt][nt][r] + bb;
        if constexpr (ACT == ACT_SILU) v = v / (1.f + __expf(-v));
        if constexpr (ACT == ACT_RELU) v = fmaxf(v, 0.f);
        size_t o = (size_t)gm * N + gn;
        if constexpr (ACT == ACT_MOE)
          C[o] += gatew[(size_t)gm * gstride] * v;   // experts run sequentially
        else
          C[o] = v;
      }
    }
}

// ---------------------------------------------------------------------------
// Fused flash attention. Per block: one (b,h), 128 query rows.
// Loop 64-key blocks: S=Q*K^T (WMMA) -> LDS, online softmax (m,l,alpha) ->
// bf16 P, O += P*V (WMMA, V staged transposed). O in registers throughout.
// ---------------------------------------------------------------------------
__global__ __launch_bounds__(256) void attn_k(const float* __restrict__ qkv,
                                              float* __restrict__ obuf) {
  constexpr int LT = 72, BQ = 128, BKV = 64, LS = 68;
  __shared__ __align__(16) bf16 Qs[BQ * LT];
  __shared__ __align__(16) bf16 Ks[BKV * LT];
  __shared__ __align__(16) bf16 Vt[CDH * LT];
  __shared__ __align__(16) bf16 Ps[BQ * LT];
  __shared__ __align__(16) float Sld[BQ * LS];
  __shared__ float mrun[BQ], lrun[BQ], alph[BQ];

  const int tid = threadIdx.x, lane = tid & 31, wid = tid >> 5;
  const int wm = (wid & 3) * 32, wn = (wid >> 2) * 32;
  const int nl = lane & 15, mr = (lane >> 4) * 8;
  const int qb = blockIdx.x * BQ, h = blockIdx.y, b = blockIdx.z;
  const size_t rs = 3 * CD;  // qkv row stride
  const float* qp = qkv + (size_t)b * CS * rs + h * CDH;
  const float* kp = qp + CD;
  const float* vp = qp + 2 * CD;

  // Stage Q block (128 x 64)
#pragma unroll
  for (int i = 0; i < 8; ++i) {
    int idx = tid + i * 256, r = idx >> 4, c = (idx & 15) << 2;
    float4 v = *(const float4*)(qp + (size_t)(qb + r) * rs + c);
    bf16* d = &Qs[r * LT + c];
    d[0] = (bf16)v.x; d[1] = (bf16)v.y; d[2] = (bf16)v.z; d[3] = (bf16)v.w;
  }
  if (tid < BQ) { mrun[tid] = -1e30f; lrun[tid] = 0.f; }

  v8f oacc[2][2];
#pragma unroll
  for (int i = 0; i < 2; ++i)
#pragma unroll
    for (int j = 0; j < 2; ++j)
#pragma unroll
      for (int r = 0; r < 8; ++r) oacc[i][j][r] = 0.f;
  __syncthreads();

  for (int kv = 0; kv < CS; kv += BKV) {
    // Stage K (row-major) and V (transposed) blocks, 64x64 each
#pragma unroll
    for (int i = 0; i < 4; ++i) {
      int idx = tid + i * 256, r = idx >> 4, c = (idx & 15) << 2;
      float4 kvv = *(const float4*)(kp + (size_t)(kv + r) * rs + c);
      bf16* d = &Ks[r * LT + c];
      d[0] = (bf16)kvv.x; d[1] = (bf16)kvv.y; d[2] = (bf16)kvv.z; d[3] = (bf16)kvv.w;
      float4 vv = *(const float4*)(vp + (size_t)(kv + r) * rs + c);
      Vt[(c + 0) * LT + r] = (bf16)vv.x;
      Vt[(c + 1) * LT + r] = (bf16)vv.y;
      Vt[(c + 2) * LT + r] = (bf16)vv.z;
      Vt[(c + 3) * LT + r] = (bf16)vv.w;
    }
    __syncthreads();

    // S = Q K^T * 1/sqrt(dh)
    v8f sacc[2][2];
#pragma unroll
    for (int i = 0; i < 2; ++i)
#pragma unroll
      for (int j = 0; j < 2; ++j)
#pragma unroll
        for (int r = 0; r < 8; ++r) sacc[i][j][r] = 0.f;
#pragma unroll
    for (int kk = 0; kk < CDH; kk += 32) {
      v16bf af[2], wf[2];
#pragma unroll
      for (int mt = 0; mt < 2; ++mt)
        af[mt] = frag_a(&Qs[(wm + mt * 16 + nl) * LT + kk], lane);
#pragma unroll
      for (int nt = 0; nt < 2; ++nt)
        wf[nt] = frag_b(&Ks[(wn + nt * 16 + nl) * LT + kk], lane);
#pragma unroll
      for (int mt = 0; mt < 2; ++mt)
#pragma unroll
        for (int nt = 0; nt < 2; ++nt)
          sacc[mt][nt] = wmma_bf16(af[mt], wf[nt], sacc[mt][nt]);
    }
#pragma unroll
    for (int mt = 0; mt < 2; ++mt)
#pragma unroll
      for (int nt = 0; nt < 2; ++nt)
#pragma unroll
        for (int r = 0; r < 8; ++r)
          Sld[(wm + mt * 16 + mr + r) * LS + wn + nt * 16 + nl] =
              sacc[mt][nt][r] * 0.125f;
    __syncthreads();

    // Online softmax, one thread per query row
    if (tid < BQ) {
      float mo = mrun[tid], mx = mo;
      const float* sr = &Sld[tid * LS];
      for (int j = 0; j < BKV; ++j) mx = fmaxf(mx, sr[j]);
      float a = __expf(mo - mx);
      float sum = 0.f;
      bf16* pr = &Ps[tid * LT];
      for (int j = 0; j < BKV; ++j) {
        float p = __expf(sr[j] - mx);
        sum += p;
        pr[j] = (bf16)p;
      }
      mrun[tid] = mx;
      lrun[tid] = lrun[tid] * a + sum;
      alph[tid] = a;
    }
    __syncthreads();

    // O = O*alpha + P*V
#pragma unroll
    for (int mt = 0; mt < 2; ++mt)
#pragma unroll
      for (int nt = 0; nt < 2; ++nt)
#pragma unroll
        for (int r = 0; r < 8; ++r)
          oacc[mt][nt][r] *= alph[wm + mt * 16 + mr + r];
#pragma unroll
    for (int kk = 0; kk < BKV; kk += 32) {
      v16bf af[2], wf[2];
#pragma unroll
      for (int mt = 0; mt < 2; ++mt)
        af[mt] = frag_a(&Ps[(wm + mt * 16 + nl) * LT + kk], lane);
#pragma unroll
      for (int nt = 0; nt < 2; ++nt)
        wf[nt] = frag_b(&Vt[(wn + nt * 16 + nl) * LT + kk], lane);
#pragma unroll
      for (int mt = 0; mt < 2; ++mt)
#pragma unroll
        for (int nt = 0; nt < 2; ++nt)
          oacc[mt][nt] = wmma_bf16(af[mt], wf[nt], oacc[mt][nt]);
    }
    __syncthreads();
  }

  // O /= l, write to [b, s, h*64 + d]
#pragma unroll
  for (int mt = 0; mt < 2; ++mt)
#pragma unroll
    for (int nt = 0; nt < 2; ++nt)
#pragma unroll
      for (int r = 0; r < 8; ++r) {
        int row = wm + mt * 16 + mr + r, col = wn + nt * 16 + nl;
        obuf[((size_t)b * CS + qb + row) * CD + h * CDH + col] =
            oacc[mt][nt][r] / lrun[row];
      }
}

// ---------------------------------------------------------------------------
// Elementwise / norm / router kernels
// ---------------------------------------------------------------------------
__global__ __launch_bounds__(256) void embed_k(const int* __restrict__ tok,
                                               const float* __restrict__ emb,
                                               float* __restrict__ x) {
  int t = blockIdx.x, c = threadIdx.x * 4;
  int id = tok[t];
  float4 v = *(const float4*)(emb + (size_t)id * CD + c);
  v.x *= 32.f; v.y *= 32.f; v.z *= 32.f; v.w *= 32.f;  // sqrt(1024)
  *(float4*)(x + (size_t)t * CD + c) = v;
}

// x = LN(x + r1) if r2==null else LN(x + 0.5*(r1+r2))
__global__ __launch_bounds__(256) void ln_k(float* __restrict__ x,
                                            const float* __restrict__ r1,
                                            const float* __restrict__ r2,
                                            const float* __restrict__ w,
                                            const float* __restrict__ bsh) {
  __shared__ float s1[256], s2[256];
  int tid = threadIdx.x;
  size_t base = (size_t)blockIdx.x * CD + tid * 4;
  float4 xv = *(const float4*)(x + base);
  float4 av = *(const float4*)(r1 + base);
  float i0, i1, i2, i3;
  if (r2) {
    float4 bv = *(const float4*)(r2 + base);
    i0 = xv.x + 0.5f * (av.x + bv.x);
    i1 = xv.y + 0.5f * (av.y + bv.y);
    i2 = xv.z + 0.5f * (av.z + bv.z);
    i3 = xv.w + 0.5f * (av.w + bv.w);
  } else {
    i0 = xv.x + av.x; i1 = xv.y + av.y; i2 = xv.z + av.z; i3 = xv.w + av.w;
  }
  s1[tid] = i0 + i1 + i2 + i3;
  s2[tid] = i0 * i0 + i1 * i1 + i2 * i2 + i3 * i3;
  __syncthreads();
  for (int st = 128; st > 0; st >>= 1) {
    if (tid < st) { s1[tid] += s1[tid + st]; s2[tid] += s2[tid + st]; }
    __syncthreads();
  }
  float mu = s1[0] * (1.f / CD);
  float inv = rsqrtf(s2[0] * (1.f / CD) - mu * mu + 1e-5f);
  int c = tid * 4;
  float4 o;
  o.x = (i0 - mu) * inv * w[c + 0] + bsh[c + 0];
  o.y = (i1 - mu) * inv * w[c + 1] + bsh[c + 1];
  o.z = (i2 - mu) * inv * w[c + 2] + bsh[c + 2];
  o.w = (i3 - mu) * inv * w[c + 3] + bsh[c + 3];
  *(float4*)(x + base) = o;
}

__global__ __launch_bounds__(256) void rms_k(const float* __restrict__ x,
                                             const float* __restrict__ w,
                                             float* __restrict__ y) {
  __shared__ float s1[256];
  int tid = threadIdx.x;
  size_t base = (size_t)blockIdx.x * CD + tid * 4;
  float4 xv = *(const float4*)(x + base);
  s1[tid] = xv.x * xv.x + xv.y * xv.y + xv.z * xv.z + xv.w * xv.w;
  __syncthreads();
  for (int st = 128; st > 0; st >>= 1) {
    if (tid < st) s1[tid] += s1[tid + st];
    __syncthreads();
  }
  float inv = rsqrtf(s1[0] * (1.f / CD) + 1e-6f);
  int c = tid * 4;
  float4 o;
  o.x = xv.x * inv * w[c + 0];
  o.y = xv.y * inv * w[c + 1];
  o.z = xv.z * inv * w[c + 2];
  o.w = xv.w * inv * w[c + 3];
  *(float4*)(y + base) = o;
}

// Router: gate scores, top-2, softmax weights -> dense_w[t, 8]
__global__ __launch_bounds__(256) void router_k(const float* __restrict__ x,
                                                const float* __restrict__ gw,
                                                const float* __restrict__ gb,
                                                float* __restrict__ dw) {
  int t = blockIdx.x * 256 + threadIdx.x;
  if (t >= MTOK) return;
  const float4* xr = (const float4*)(x + (size_t)t * CD);
  float s[CE];
  for (int e = 0; e < CE; ++e) {
    const float4* wr = (const float4*)(gw + (size_t)e * CD);
    float acc = 0.f;
    for (int i = 0; i < CD / 4; ++i) {
      float4 a = xr[i], ww = wr[i];
      acc += a.x * ww.x + a.y * ww.y + a.z * ww.z + a.w * ww.w;
    }
    s[e] = acc + gb[e];
  }
  int i0 = 0;
  for (int e = 1; e < CE; ++e) if (s[e] > s[i0]) i0 = e;
  int i1 = (i0 == 0) ? 1 : 0;
  for (int e = 0; e < CE; ++e)
    if (e != i0 && s[e] > s[i1]) i1 = e;
  float p0 = 1.f / (1.f + __expf(s[i1] - s[i0]));
  float p1 = 1.f - p0;
  for (int e = 0; e < CE; ++e)
    dw[(size_t)t * CE + e] = (e == i0) ? p0 : ((e == i1) ? p1 : 0.f);
}

__global__ __launch_bounds__(256) void mul_k(float* __restrict__ a,
                                             const float* __restrict__ b,
                                             int n4) {
  int i = blockIdx.x * 256 + threadIdx.x;
  if (i >= n4) return;
  float4 av = ((const float4*)a)[i];
  float4 bv = ((const float4*)b)[i];
  av.x *= bv.x; av.y *= bv.y; av.z *= bv.z; av.w *= bv.w;
  ((float4*)a)[i] = av;
}

__global__ __launch_bounds__(256) void zero_k(float* __restrict__ p, int n4) {
  int i = blockIdx.x * 256 + threadIdx.x;
  if (i >= n4) return;
  ((float4*)p)[i] = make_float4(0.f, 0.f, 0.f, 0.f);
}

// ---------------------------------------------------------------------------
// Host-side orchestration (all on `stream`, graph-capture safe, deterministic)
// ---------------------------------------------------------------------------
extern "C" void kernel_launch(void* const* d_in, const int* in_sizes, int n_in,
                              void* d_out, int out_size, void* d_ws, size_t ws_size,
                              hipStream_t stream) {
  (void)in_sizes; (void)n_in; (void)out_size; (void)ws_size;
  const int*   tokens     = (const int*)d_in[0];
  const float* emb        = (const float*)d_in[1];
  const float* in_proj_w  = (const float*)d_in[2];
  const float* in_proj_b  = (const float*)d_in[3];
  const float* out_proj_w = (const float*)d_in[4];
  const float* out_proj_b = (const float*)d_in[5];
  const float* ln1_w      = (const float*)d_in[6];
  const float* ln1_b      = (const float*)d_in[7];
  const float* ln2_w      = (const float*)d_in[8];
  const float* ln2_b      = (const float*)d_in[9];
  const float* ds_gate_w  = (const float*)d_in[10];
  const float* ds_up_w    = (const float*)d_in[11];
  const float* ds_down_w  = (const float*)d_in[12];
  const float* moe_w1     = (const float*)d_in[13];
  const float* moe_b1     = (const float*)d_in[14];
  const float* moe_w2     = (const float*)d_in[15];
  const float* moe_b2     = (const float*)d_in[16];
  const float* gate_w     = (const float*)d_in[17];
  const float* gate_b     = (const float*)d_in[18];
  const float* rms_w      = (const float*)d_in[19];

  float* ws = (float*)d_ws;
  size_t off = 0;
  auto alloc = [&](size_t n) { float* p = ws + off; off += n; return p; };
  float* xbuf  = alloc((size_t)MTOK * CD);       // activations
  float* qkvb  = alloc((size_t)MTOK * 3 * CD);   // packed qkv
  float* obuf  = alloc((size_t)MTOK * CD);       // attention output
  float* aout  = alloc((size_t)MTOK * CD);       // out_proj output
  float* t1    = alloc((size_t)MTOK * CF);       // silu(gate)
  float* t2    = alloc((size_t)MTOK * CF);       // up
  float* dsout = alloc((size_t)MTOK * CD);       // DS-MLP output
  float* hbuf  = alloc((size_t)MTOK * CF);       // expert hidden (reused per e)
  float* moeb  = alloc((size_t)MTOK * CD);       // MoE accumulator
  float* dw    = alloc((size_t)MTOK * CE);       // router dense weights
  float* xn    = alloc((size_t)MTOK * CD);       // rms-normed final

  dim3 blk(256);

  embed_k<<<MTOK, blk, 0, stream>>>(tokens, emb, xbuf);

  for (int l = 0; l < 2; ++l) {
    // QKV projection
    gemm_k<ACT_NONE><<<dim3(3 * CD / 128, MTOK / 128), blk, 0, stream>>>(
        xbuf, in_proj_w + (size_t)l * 3 * CD * CD, in_proj_b + (size_t)l * 3 * CD,
        qkvb, MTOK, 3 * CD, CD, nullptr, 0);
    // Fused flash attention
    attn_k<<<dim3(CS / 128, CH, CB), blk, 0, stream>>>(qkvb, obuf);
    // Output projection
    gemm_k<ACT_NONE><<<dim3(CD / 128, MTOK / 128), blk, 0, stream>>>(
        obuf, out_proj_w + (size_t)l * CD * CD, out_proj_b + (size_t)l * CD,
        aout, MTOK, CD, CD, nullptr, 0);
    // x = LN(x + attn_out)
    ln_k<<<MTOK, blk, 0, stream>>>(xbuf, aout, nullptr,
                                   ln1_w + (size_t)l * CD, ln1_b + (size_t)l * CD);
    // DS-MLP: down(silu(gate(x)) * up(x))
    gemm_k<ACT_SILU><<<dim3(CF / 128, MTOK / 128), blk, 0, stream>>>(
        xbuf, ds_gate_w + (size_t)l * CF * CD, nullptr, t1, MTOK, CF, CD, nullptr, 0);
    gemm_k<ACT_NONE><<<dim3(CF / 128, MTOK / 128), blk, 0, stream>>>(
        xbuf, ds_up_w + (size_t)l * CF * CD, nullptr, t2, MTOK, CF, CD, nullptr, 0);
    mul_k<<<(MTOK * CF / 4 + 255) / 256, blk, 0, stream>>>(t1, t2, MTOK * CF / 4);
    gemm_k<ACT_NONE><<<dim3(CD / 128, MTOK / 128), blk, 0, stream>>>(
        t1, ds_down_w + (size_t)l * CD * CF, nullptr, dsout, MTOK, CD, CF, nullptr, 0);
    // Router + MoE (per-expert GEMMs; w2 fuses weighted accumulate)
    router_k<<<MTOK / 256, blk, 0, stream>>>(xbuf, gate_w + (size_t)l * CE * CD,
                                             gate_b + (size_t)l * CE, dw);
    zero_k<<<(MTOK * CD / 4 + 255) / 256, blk, 0, stream>>>(moeb, MTOK * CD / 4);
    for (int e = 0; e < CE; ++e) {
      size_t we = (size_t)(l * CE + e);
      gemm_k<ACT_RELU><<<dim3(CF / 128, MTOK / 128), blk, 0, stream>>>(
          xbuf, moe_w1 + we * CF * CD, moe_b1 + we * CF, hbuf, MTOK, CF, CD,
          nullptr, 0);
      gemm_k<ACT_MOE><<<dim3(CD / 128, MTOK / 128), blk, 0, stream>>>(
          hbuf, moe_w2 + we * CD * CF, moe_b2 + we * CD, moeb, MTOK, CD, CF,
          dw + e, CE);
    }
    // x = LN(x + 0.5*(ds + moe))
    ln_k<<<MTOK, blk, 0, stream>>>(xbuf, dsout, moeb,
                                   ln2_w + (size_t)l * CD, ln2_b + (size_t)l * CD);
  }

  // Final RMSNorm + tied lm_head
  rms_k<<<MTOK, blk, 0, stream>>>(xbuf, rms_w, xn);
  gemm_k<ACT_NONE><<<dim3(CV / 128, MTOK / 128), blk, 0, stream>>>(
      xn, emb, nullptr, (float*)d_out, MTOK, CV, CD, nullptr, 0);
}
